// pointnet2_cls_ssg_81217831567444
// MI455X (gfx1250) — compile-verified, matrix operand order checked by reading, not run on hardware
//
#include <hip/hip_runtime.h>
#include <hip/hip_bf16.h>

typedef __attribute__((ext_vector_type(16))) _Float16     v16h;
typedef __attribute__((ext_vector_type(8)))  float        v8f;
typedef __attribute__((ext_vector_type(4)))  unsigned int v4u;

// ---------------------------------------------------------------------------
// Farthest point sampling: one block per batch, sequential M steps with an
// LDS argmax tree reduction (matches jnp.argmax first-occurrence tie-break).
// ---------------------------------------------------------------------------
__global__ void fps_kernel(const float* __restrict__ xyz, int N, int M,
                           int* __restrict__ out_idx) {
  const int b = blockIdx.x;
  const int t = threadIdx.x;                 // 256 threads
  const float* p = xyz + (size_t)b * N * 3;
  const int PT = (N + 255) / 256;            // points per thread (<=4)

  float px[4], py[4], pz[4], dist[4];
  for (int i = 0; i < 4; ++i) {
    int n = t + i * 256;
    if (i < PT && n < N) {
      px[i] = p[n * 3 + 0]; py[i] = p[n * 3 + 1]; pz[i] = p[n * 3 + 2];
      dist[i] = 1e10f;
    } else {
      dist[i] = -1e30f;
    }
  }

  __shared__ float smax[256];
  __shared__ int   simx[256];
  __shared__ float cx, cy, cz;

  int far = 0;
  for (int it = 0; it < M; ++it) {
    if (t == 0) {
      out_idx[(size_t)b * M + it] = far;
      cx = p[far * 3 + 0]; cy = p[far * 3 + 1]; cz = p[far * 3 + 2];
    }
    __syncthreads();
    float bx = cx, by = cy, bz = cz;
    float best = -1e30f; int bidx = 0x7fffffff;
    for (int i = 0; i < 4; ++i) {
      int n = t + i * 256;
      if (i < PT && n < N) {
        float dx = px[i] - bx, dy = py[i] - by, dz = pz[i] - bz;
        float d = dx * dx + dy * dy + dz * dz;
        dist[i] = fminf(dist[i], d);
        if (dist[i] > best) { best = dist[i]; bidx = n; }
      }
    }
    smax[t] = best; simx[t] = bidx;
    __syncthreads();
    for (int s = 128; s > 0; s >>= 1) {
      if (t < s) {
        if (smax[t + s] > smax[t] ||
            (smax[t + s] == smax[t] && simx[t + s] < simx[t])) {
          smax[t] = smax[t + s]; simx[t] = simx[t + s];
        }
      }
      __syncthreads();
    }
    far = simx[0];
    __syncthreads();
  }
}

// Gather centroid coordinates: new_xyz[b,m,:] = xyz[b, fi[b,m], :]
__global__ void gather_xyz_kernel(const float* __restrict__ xyz,
                                  const int* __restrict__ fi,
                                  int N, int M, int total,
                                  float* __restrict__ out) {
  int idx = blockIdx.x * blockDim.x + threadIdx.x;
  if (idx >= total) return;                  // total = B*M
  int b = idx / M;
  int n = fi[idx];
  const float* s = xyz + ((size_t)b * N + n) * 3;
  out[(size_t)idx * 3 + 0] = s[0];
  out[(size_t)idx * 3 + 1] = s[1];
  out[(size_t)idx * 3 + 2] = s[2];
}

// Ball query: first K indices (ascending index order == first hits of the
// scan) within radius; pad with first hit (reference semantics).
__global__ void ball_query_kernel(const float* __restrict__ xyz,
                                  const float* __restrict__ new_xyz,
                                  int N, int M, int K, float r2, int total,
                                  int* __restrict__ out) {
  int idx = blockIdx.x * blockDim.x + threadIdx.x;
  if (idx >= total) return;                  // total = B*M
  int b = idx / M;
  const float* p = xyz + (size_t)b * N * 3;
  float qx = new_xyz[(size_t)idx * 3 + 0];
  float qy = new_xyz[(size_t)idx * 3 + 1];
  float qz = new_xyz[(size_t)idx * 3 + 2];
  int* o = out + (size_t)idx * K;
  int cnt = 0, first = 0;
  for (int n = 0; n < N; ++n) {
    float dx = p[n * 3 + 0] - qx, dy = p[n * 3 + 1] - qy, dz = p[n * 3 + 2] - qz;
    float d = dx * dx + dy * dy + dz * dz;
    if (d <= r2) {
      if (cnt == 0) first = n;
      o[cnt] = n;
      if (++cnt >= K) break;
    }
  }
  if (cnt == 0) { o[0] = 0; first = 0; cnt = 1; }
  for (int k = cnt; k < K; ++k) o[k] = first;
}

// Group: out[(b,m,k), 0:3] = xyz[gidx]-new_xyz ; [3:3+Cf] = feats[gidx]; pad 0.
__global__ void group_kernel(const float* __restrict__ xyz,
                             const float* __restrict__ new_xyz,
                             const int* __restrict__ gidx,
                             const float* __restrict__ feat_f,
                             const _Float16* __restrict__ feat_h,
                             int N, int M, int K, int Cf, int CinPad, int total,
                             _Float16* __restrict__ out) {
  int tid = blockIdx.x * blockDim.x + threadIdx.x;
  if (tid >= total) return;                  // total = B*M*K
  int m = (tid / K) % M;
  int b = tid / (K * M);
  int n = gidx[tid];
  _Float16* o = out + (size_t)tid * CinPad;
  const float* c = new_xyz + ((size_t)b * M + m) * 3;
  const float* s = xyz + ((size_t)b * N + n) * 3;
  o[0] = (_Float16)(s[0] - c[0]);
  o[1] = (_Float16)(s[1] - c[1]);
  o[2] = (_Float16)(s[2] - c[2]);
  if (feat_f) {
    const float* f = feat_f + ((size_t)b * N + n) * Cf;
    for (int cc = 0; cc < Cf; ++cc) o[3 + cc] = (_Float16)f[cc];
  } else {
    const _Float16* f = feat_h + ((size_t)b * N + n) * Cf;
    for (int cc = 0; cc < Cf; ++cc) o[3 + cc] = f[cc];
  }
  for (int cc = 3 + Cf; cc < CinPad; ++cc) o[cc] = (_Float16)0.0f;
}

// group_all: out[(b,n), 0:3] = x2 (raw coords), [3:3+Cf] = f2, pad 0.
__global__ void group_all_kernel(const float* __restrict__ x2,
                                 const _Float16* __restrict__ f2,
                                 int Cf, int CinPad, int total,
                                 _Float16* __restrict__ out) {
  int tid = blockIdx.x * blockDim.x + threadIdx.x;
  if (tid >= total) return;                  // total = B*Npts
  _Float16* o = out + (size_t)tid * CinPad;
  o[0] = (_Float16)x2[(size_t)tid * 3 + 0];
  o[1] = (_Float16)x2[(size_t)tid * 3 + 1];
  o[2] = (_Float16)x2[(size_t)tid * 3 + 2];
  const _Float16* f = f2 + (size_t)tid * Cf;
  for (int cc = 0; cc < Cf; ++cc) o[3 + cc] = f[cc];
  for (int cc = 3 + Cf; cc < CinPad; ++cc) o[cc] = (_Float16)0.0f;
}

// ---------------------------------------------------------------------------
// Pack f32 weight [Cin,Cout] into WMMA-B fragment-ready f16 layout:
// dword index = ((kb32*tilesN + tn)*32 + lane)*8 + v, holding the (kk,kk+1)
// f16 pair for kk = kb32*32 + (v>>2)*16 + (lane>>4)*8 + (v&3)*2,
// n = tn*16 + (lane&15). GEMM B-load then = two global_load_b128 per frag.
// ---------------------------------------------------------------------------
__global__ void pack_w_kernel(const float* __restrict__ w,
                              int Cin, int Cout, int CinPad, int CoutPad,
                              _Float16* __restrict__ out) {
  int idx = blockIdx.x * blockDim.x + threadIdx.x;
  int total = (CinPad * CoutPad) >> 1;       // dword count
  if (idx >= total) return;
  int tilesN = CoutPad >> 4;
  int v    = idx & 7;
  int lane = (idx >> 3) & 31;
  int tn   = (idx >> 8) % tilesN;
  int kb32 = (idx >> 8) / tilesN;
  int kk = kb32 * 32 + ((v >> 2) << 4) + ((lane >> 4) << 3) + ((v & 3) << 1);
  int n  = (tn << 4) + (lane & 15);
  float f0 = (kk + 0 < Cin && n < Cout) ? w[(size_t)(kk + 0) * Cout + n] : 0.0f;
  float f1 = (kk + 1 < Cin && n < Cout) ? w[(size_t)(kk + 1) * Cout + n] : 0.0f;
  out[(size_t)idx * 2 + 0] = (_Float16)f0;
  out[(size_t)idx * 2 + 1] = (_Float16)f1;
}

// ---------------------------------------------------------------------------
// WMMA GEMM: Y[R,CoutPad] = relu(X[R,CinPad] @ W + bias), W pre-packed.
// One wave32 owns a 32x32 output tile (2x2 sub-tiles): 2 A frags + 2 B frags
// feed 4 independent v_wmma_f32_16x16x32_f16 chains per K-step. All fragment
// loads are b128 (A halves are contiguous per the ISA 16-bit 16x32 layout).
// ---------------------------------------------------------------------------
__global__ void mlp_wmma_kernel(const _Float16* __restrict__ X,
                                const _Float16* __restrict__ Wp,
                                const float* __restrict__ bias,
                                int R, int CinPad, int CoutPad, int CoutStore,
                                int reluFlag,
                                _Float16* __restrict__ Yh,
                                float* __restrict__ Yf) {
  const int wave = (blockIdx.x * blockDim.x + threadIdx.x) >> 5;
  const int lane = threadIdx.x & 31;
  const int colT2  = CoutPad >> 5;
  const int nTiles = (R >> 5) * colT2;
  if (wave >= nTiles) return;                // wave-uniform: EXEC stays all-1s

  const int tm = wave / colT2, tn2 = wave % colT2;
  const int row0 = tm << 5, col0 = tn2 << 5;
  const int lrow = lane & 15, half = lane >> 4;
  const int tilesN = CoutPad >> 4;

  v8f acc00 = {}, acc01 = {}, acc10 = {}, acc11 = {};
  const _Float16* a0p = X + (size_t)(row0 + lrow) * CinPad + half * 8;
  const _Float16* a1p = a0p + (size_t)16 * CinPad;

  union Frag { v4u q[2]; v16h v; };

  for (int kb = 0; kb < CinPad; kb += 32) {
    Frag a0, a1, b0, b1;
    a0.q[0] = *(const v4u*)(a0p + kb);
    a0.q[1] = *(const v4u*)(a0p + kb + 16);
    a1.q[0] = *(const v4u*)(a1p + kb);
    a1.q[1] = *(const v4u*)(a1p + kb + 16);
    const _Float16* bp =
        Wp + (((size_t)(kb >> 5) * tilesN + 2 * tn2) * 32 + lane) * 16;
    b0.q[0] = *(const v4u*)(bp);
    b0.q[1] = *(const v4u*)(bp + 8);
    b1.q[0] = *(const v4u*)(bp + 512);       // next 16-col tile block
    b1.q[1] = *(const v4u*)(bp + 520);
    __builtin_prefetch(a0p + kb + 32, 0, 1); // next A K-block -> global_prefetch
    acc00 = __builtin_amdgcn_wmma_f32_16x16x32_f16(false, a0.v, false, b0.v,
                                                   (short)0, acc00, false, false);
    acc01 = __builtin_amdgcn_wmma_f32_16x16x32_f16(false, a0.v, false, b1.v,
                                                   (short)0, acc01, false, false);
    acc10 = __builtin_amdgcn_wmma_f32_16x16x32_f16(false, a1.v, false, b0.v,
                                                   (short)0, acc10, false, false);
    acc11 = __builtin_amdgcn_wmma_f32_16x16x32_f16(false, a1.v, false, b1.v,
                                                   (short)0, acc11, false, false);
  }

  const int n0 = col0 + lrow, n1 = n0 + 16;
  const float bv0 = (bias != nullptr && n0 < CoutStore) ? bias[n0] : 0.0f;
  const float bv1 = (bias != nullptr && n1 < CoutStore) ? bias[n1] : 0.0f;

  auto store_tile = [&](v8f acc, int rbase, int n, float bv) {
#pragma unroll
    for (int r = 0; r < 8; ++r) {
      const int m = rbase + half * 8 + r;    // ISA C/D layout
      float val = acc[r] + bv;
      if (reluFlag) val = fmaxf(val, 0.0f);
      if (n < CoutStore) {
        if (Yh) Yh[(size_t)m * CoutPad + n] = (_Float16)val;
        if (Yf) Yf[(size_t)m * CoutStore + n] = val;
      }
    }
  };
  store_tile(acc00, row0 +  0, n0, bv0);
  store_tile(acc01, row0 +  0, n1, bv1);
  store_tile(acc10, row0 + 16, n0, bv0);
  store_tile(acc11, row0 + 16, n1, bv1);
}

// Max over K neighbors: [BM*K, C] -> [BM, C]
__global__ void max_k_kernel(const _Float16* __restrict__ y,
                             int K, int C, int total,
                             _Float16* __restrict__ out) {
  int idx = blockIdx.x * blockDim.x + threadIdx.x;
  if (idx >= total) return;                  // total = BM*C
  int c = idx % C, g = idx / C;
  float best = -1e30f;
  for (int k = 0; k < K; ++k)
    best = fmaxf(best, (float)y[((size_t)g * K + k) * C + c]);
  out[idx] = (_Float16)best;
}

// ---------------------------------------------------------------------------
static inline int cdiv(int a, int b) { return (a + b - 1) / b; }

extern "C" void kernel_launch(void* const* d_in, const int* in_sizes, int n_in,
                              void* d_out, int out_size, void* d_ws, size_t ws_size,
                              hipStream_t stream) {
  (void)in_sizes; (void)n_in; (void)out_size; (void)ws_size;
  const int B = 32, N = 1024;

  const float* xyz    = (const float*)d_in[0];
  const float* points = (const float*)d_in[1];
  const float* w[9]; const float* bb[9];
  for (int i = 0; i < 9; ++i) { w[i] = (const float*)d_in[2 + 2 * i]; bb[i] = (const float*)d_in[3 + 2 * i]; }
  const float* lin1_w = (const float*)d_in[20];
  const float* lin2_w = (const float*)d_in[21];
  const float* cls_w  = (const float*)d_in[22];
  const float* cls_b  = (const float*)d_in[23];
  float* out = (float*)d_out;

  // Bump allocator over d_ws
  char* ws = (char*)d_ws; size_t off = 0;
  auto alloc = [&](size_t bytes) -> void* {
    off = (off + 255) & ~(size_t)255;
    void* p = ws + off; off += bytes; return p;
  };

  int*      fi1 = (int*)alloc((size_t)B * 512 * 4);
  float*    nx1 = (float*)alloc((size_t)B * 512 * 3 * 4);
  int*      gi1 = (int*)alloc((size_t)B * 512 * 32 * 4);
  int*      fi2 = (int*)alloc((size_t)B * 128 * 4);
  float*    nx2 = (float*)alloc((size_t)B * 128 * 3 * 4);
  int*      gi2 = (int*)alloc((size_t)B * 128 * 64 * 4);
  _Float16* f1  = (_Float16*)alloc((size_t)B * 512 * 128 * 2);
  _Float16* f2  = (_Float16*)alloc((size_t)B * 128 * 256 * 2);
  _Float16* f3  = (_Float16*)alloc((size_t)B * 1024 * 2);
  _Float16* fc1 = (_Float16*)alloc((size_t)B * 512 * 2);
  _Float16* fc2 = (_Float16*)alloc((size_t)B * 256 * 2);

  // Padded, fragment-packed f16 weights: {Cin,Cout,CinPad,CoutPad}
  const int wd[12][4] = {
    {6, 64, 32, 64}, {64, 64, 64, 64}, {64, 128, 64, 128},            // SA1
    {131, 128, 160, 128}, {128, 128, 128, 128}, {128, 256, 128, 256}, // SA2
    {259, 256, 288, 256}, {256, 512, 256, 512}, {512, 1024, 512, 1024}, // SA3
    {1024, 512, 1024, 512}, {512, 256, 512, 256}, {256, 40, 256, 64}  // FC
  };
  const float* wsrc[12] = { w[0], w[1], w[2], w[3], w[4], w[5], w[6], w[7], w[8],
                            lin1_w, lin2_w, cls_w };
  _Float16* Wh[12];
  for (int i = 0; i < 12; ++i) {
    int tot = wd[i][2] * wd[i][3];
    Wh[i] = (_Float16*)alloc((size_t)tot * 2);
    pack_w_kernel<<<cdiv(tot / 2, 256), 256, 0, stream>>>(
        wsrc[i], wd[i][0], wd[i][1], wd[i][2], wd[i][3], Wh[i]);
  }

  _Float16* bufA = (_Float16*)alloc((size_t)262144 * 160 * 2);   // 80 MB peak (g2)
  _Float16* bufB = (_Float16*)alloc((size_t)524288 * 128 * 2);   // 128 MB peak (SA1 L3)

  auto mlp = [&](const _Float16* X, int li, const float* bias, int R,
                 int relu, _Float16* Yh, float* Yf, int coutStore) {
    int nTiles = (R >> 5) * (wd[li][3] >> 5);
    mlp_wmma_kernel<<<cdiv(nTiles, 4), 128, 0, stream>>>(
        X, Wh[li], bias, R, wd[li][2], wd[li][3], coutStore, relu, Yh, Yf);
  };

  // ---------------- SA1: FPS(512) -> ball(0.2,K=32) -> MLP -> max ----------
  fps_kernel<<<B, 256, 0, stream>>>(xyz, N, 512, fi1);
  gather_xyz_kernel<<<cdiv(B * 512, 256), 256, 0, stream>>>(xyz, fi1, N, 512, B * 512, nx1);
  ball_query_kernel<<<cdiv(B * 512, 256), 256, 0, stream>>>(xyz, nx1, N, 512, 32,
                                                            0.2f * 0.2f, B * 512, gi1);
  group_kernel<<<cdiv(B * 512 * 32, 256), 256, 0, stream>>>(
      xyz, nx1, gi1, points, nullptr, N, 512, 32, 3, 32, B * 512 * 32, bufA);
  const int R1 = B * 512 * 32;
  mlp(bufA, 0, bb[0], R1, 1, bufB, nullptr, 64);
  mlp(bufB, 1, bb[1], R1, 1, bufA, nullptr, 64);
  mlp(bufA, 2, bb[2], R1, 1, bufB, nullptr, 128);
  max_k_kernel<<<cdiv(B * 512 * 128, 256), 256, 0, stream>>>(bufB, 32, 128, B * 512 * 128, f1);

  // ---------------- SA2: FPS(128) -> ball(0.4,K=64) -> MLP -> max ----------
  fps_kernel<<<B, 256, 0, stream>>>(nx1, 512, 128, fi2);
  gather_xyz_kernel<<<cdiv(B * 128, 256), 256, 0, stream>>>(nx1, fi2, 512, 128, B * 128, nx2);
  ball_query_kernel<<<cdiv(B * 128, 256), 256, 0, stream>>>(nx1, nx2, 512, 128, 64,
                                                            0.4f * 0.4f, B * 128, gi2);
  group_kernel<<<cdiv(B * 128 * 64, 256), 256, 0, stream>>>(
      nx1, nx2, gi2, nullptr, f1, 512, 128, 64, 128, 160, B * 128 * 64, bufA);
  const int R2 = B * 128 * 64;
  mlp(bufA, 3, bb[3], R2, 1, bufB, nullptr, 128);
  mlp(bufB, 4, bb[4], R2, 1, bufA, nullptr, 128);
  mlp(bufA, 5, bb[5], R2, 1, bufB, nullptr, 256);
  max_k_kernel<<<cdiv(B * 128 * 256, 256), 256, 0, stream>>>(bufB, 64, 256, B * 128 * 256, f2);

  // ---------------- SA3: group_all over 128 pts -> MLP -> max --------------
  group_all_kernel<<<cdiv(B * 128, 256), 256, 0, stream>>>(nx2, f2, 256, 288, B * 128, bufA);
  const int R3 = B * 128;
  mlp(bufA, 6, bb[6], R3, 1, bufB, nullptr, 256);
  mlp(bufB, 7, bb[7], R3, 1, bufA, nullptr, 512);
  mlp(bufA, 8, bb[8], R3, 1, bufB, nullptr, 1024);
  max_k_kernel<<<cdiv(B * 1024, 256), 256, 0, stream>>>(bufB, 128, 1024, B * 1024, f3);

  // ---------------- FC head ------------------------------------------------
  mlp(f3, 9, nullptr, B, 1, fc1, nullptr, 512);
  mlp(fc1, 10, nullptr, B, 1, fc2, nullptr, 256);
  mlp(fc2, 11, cls_b, B, 0, nullptr, out, 40);   // f32 output [32,40]
}
